// DGCNN_Model_2628519985497
// MI455X (gfx1250) — compile-verified, matrix operand order checked by reading, not run on hardware
//
#include <hip/hip_runtime.h>
#include <hip/hip_bf16.h>

#define BATCH 8
#define NPTS  2048
#define KNN   20

typedef __attribute__((ext_vector_type(16))) __bf16 v16bf;
typedef __attribute__((ext_vector_type(8)))  float  v8f;
typedef __attribute__((ext_vector_type(4)))  int    v4i;

#ifndef __has_builtin
#define __has_builtin(x) 0
#endif
#if defined(__HIP_DEVICE_COMPILE__) && \
    __has_builtin(__builtin_amdgcn_global_load_async_to_lds_b128) && \
    __has_builtin(__builtin_amdgcn_s_wait_asynccnt)
#define HAVE_ASYNC_LDS 1
#else
#define HAVE_ASYNC_LDS 0
#endif

#if HAVE_ASYNC_LDS
typedef __attribute__((address_space(1))) v4i* gv4ip;   // global int4*
typedef __attribute__((address_space(3))) v4i* lv4ip;   // LDS int4*
#endif

__device__ __forceinline__ void async_copy16(const void* g, void* l) {
#if HAVE_ASYNC_LDS
  __builtin_amdgcn_global_load_async_to_lds_b128((gv4ip)g, (lv4ip)l, 0, 0);
#else
  (void)g; (void)l;
#endif
}
__device__ __forceinline__ void async_wait() {
#if HAVE_ASYNC_LDS
  __builtin_amdgcn_s_wait_asynccnt(0);
#endif
}

// Generic element loader: inputs are either strided f32 (layer-1 point cloud)
// or contiguous bf16 (xcat activations).
__device__ __forceinline__ float ldE(const void* p, int idx, int bf) {
  return bf ? (float)((const __bf16*)p)[idx] : ((const float*)p)[idx];
}

// ---------------------------------------------------------------------------
// Squared norms per point (ranking uses 2*inner - |xc|^2; -|xr|^2 is a
// row-constant and drops out of top-k).
// ---------------------------------------------------------------------------
__global__ void sqnorm_kernel(const void* __restrict__ in, int bf,
                              int bs, int ns, int cs, int Cin,
                              float* __restrict__ xx) {
  int i = blockIdx.x * blockDim.x + threadIdx.x;
  if (i >= BATCH * NPTS) return;
  int b = i / NPTS, n = i % NPTS;
  int base = b * bs + n * ns;
  float s = 0.f;
  for (int c = 0; c < Cin; ++c) { float v = ldE(in, base + c * cs, bf); s += v * v; }
  xx[i] = s;
}

// ---------------------------------------------------------------------------
// Fused distance-GEMM (bf16 WMMA) + streaming top-20 selection.
// One wave / 16 query rows. Candidate tiles double-buffered in LDS via async
// global->LDS b128 copies: issue tile t+1, WMMA on tile t while DMA in flight.
// Distance tiles never reach HBM.
// ---------------------------------------------------------------------------
__global__ __launch_bounds__(32)
void knn_topk_kernel(const void* __restrict__ in, int bf,
                     int bs, int ns, int cs, int Cin, int CP,
                     const float* __restrict__ xx,
                     int* __restrict__ idx_out) {
  __shared__ __align__(16) __bf16 qa[16][128];
  __shared__ __align__(16) __bf16 cb[2][16][128];
  __shared__ float dtile[16][16];
  __shared__ float tvals[16][KNN];
  __shared__ int   tidx[16][KNN];

  int lane = threadIdx.x;
  int b    = blockIdx.x / (NPTS / 16);
  int row0 = (blockIdx.x % (NPTS / 16)) * 16;
  int base = b * bs;
  const float* xxb = xx + b * NPTS;
  int r = lane & 15, half = lane >> 4;
  int cpsh = __builtin_ctz(CP);             // CP is a power of two
  const int NT = NPTS / 16;

#if HAVE_ASYNC_LDS
  bool use_async = bf && (CP == Cin);
  int rc = Cin >> 3;                        // 16B chunks per bf16 row
  int rcsh = __builtin_ctz(rc | 1);
  if (use_async) {
    for (int ch = lane; ch < (16 << rcsh); ch += 32) {   // queries
      int m = ch >> rcsh, c0 = (ch & (rc - 1)) << 3;
      async_copy16((const __bf16*)in + base + (row0 + m) * ns + c0, &qa[m][c0]);
    }
    for (int ch = lane; ch < (16 << rcsh); ch += 32) {   // candidate tile 0
      int m = ch >> rcsh, c0 = (ch & (rc - 1)) << 3;
      async_copy16((const __bf16*)in + base + m * ns + c0, &cb[0][m][c0]);
    }
  } else
#endif
  {
    for (int i = lane; i < 16 * CP; i += 32) {
      int m = i >> cpsh, c = i & (CP - 1);
      float v = (c < Cin) ? ldE(in, base + (row0 + m) * ns + c * cs, bf) : 0.f;
      qa[m][c] = (__bf16)v;
    }
  }

  float minv = -3.0e38f;
  int   minp = 0;
  if (lane < 16)
    for (int j = 0; j < KNN; ++j) { tvals[r][j] = -3.0e38f; tidx[r][j] = 0; }

  for (int t = 0; t < NT; ++t) {
    int cand0 = t * 16;
    const __bf16 (*cbp)[128];
#if HAVE_ASYNC_LDS
    if (use_async) {
      async_wait();                 // tile t (and queries on t==0) resident
      __syncthreads();              // prior readers of buffer (t+1)&1 done
      if (t + 1 < NT) {
        int nb = (t + 1) & 1, nc0 = cand0 + 16;
        for (int ch = lane; ch < (16 << rcsh); ch += 32) {
          int m = ch >> rcsh, c0 = (ch & (rc - 1)) << 3;
          async_copy16((const __bf16*)in + base + (nc0 + m) * ns + c0,
                       &cb[nb][m][c0]);
        }
      }
      cbp = cb[t & 1];
    } else
#endif
    {
      __syncthreads();
      for (int i = lane; i < 16 * CP; i += 32) {
        int m = i >> cpsh, c = i & (CP - 1);
        float v = (c < Cin) ? ldE(in, base + (cand0 + m) * ns + c * cs, bf) : 0.f;
        cb[0][m][c] = (__bf16)v;
      }
      if (t + 1 < NT) {             // prefetch next tile (global_prefetch_b8)
        const char* np = bf ? (const char*)((const __bf16*)in + base + (cand0 + 16 + r) * ns)
                            : (const char*)((const float*)in + base + (cand0 + 16 + r) * ns);
        __builtin_prefetch(np, 0, 1);
      }
      __syncthreads();
      cbp = cb[0];
    }

    v8f acc = {};
    int m = lane & 15;
    for (int kc = 0; kc < CP; kc += 32) {
      v16bf a, w;
#pragma unroll
      for (int j = 0; j < 8; ++j) {
        // 16-bit fragment layout: VGPR j holds K={kb,kb+1},
        // kb = (j>=4?16:0) + (j%4)*2 + 8*(lane>=16)
        int kb = kc + ((j >> 2) << 4) + ((j & 3) << 1) + (half << 3);
        a[2 * j]     = qa[m][kb];
        a[2 * j + 1] = qa[m][kb + 1];
        w[2 * j]     = cbp[m][kb];
        w[2 * j + 1] = cbp[m][kb + 1];
      }
      acc = __builtin_amdgcn_wmma_f32_16x16x32_bf16(false, a, false, w,
                                                    (short)0, acc, false, false);
    }
    float xc = xxb[cand0 + m];
#pragma unroll
    for (int j = 0; j < 8; ++j)
      dtile[half * 8 + j][m] = 2.f * acc[j] - xc;
    __syncthreads();

    if (lane < 16) {
      for (int c = 0; c < 16; ++c) {
        float v = dtile[r][c];
        if (v > minv) {
          tvals[r][minp] = v; tidx[r][minp] = cand0 + c;
          float mv = tvals[r][0]; int mp = 0;
          for (int j = 1; j < KNN; ++j)
            if (tvals[r][j] < mv) { mv = tvals[r][j]; mp = j; }
          minv = mv; minp = mp;
        }
      }
    }
  }
  if (lane < 16) {
    int* o = idx_out + (b * NPTS + row0 + r) * KNN;
    for (int j = 0; j < KNN; ++j) o[j] = tidx[r][j];
  }
}

// ---------------------------------------------------------------------------
// EdgeConv: gather 20 neighbors (padded to 32 rows by duplicating neighbor 0 —
// max over k unchanged), bf16 WMMA over [nbr-ctr | ctr], affine+leaky-relu,
// max over k. Weights staged per 16-col tile in LDS as bf16 so both WMMA
// operands come from ds_load_b128. Output written to its bf16 xcat slice.
// ---------------------------------------------------------------------------
__global__ __launch_bounds__(32)
void edge_conv_kernel(const void* __restrict__ in, int bf,
                      int bs, int ns, int cs, int Cin, int C2P, int Oc,
                      const float* __restrict__ w,
                      const float* __restrict__ sc, const float* __restrict__ bi,
                      const int* __restrict__ idx,
                      __bf16* __restrict__ xcat, int out_off) {
  __shared__ __align__(16) __bf16 feat[32][256];
  __shared__ __align__(16) __bf16 wtile[16][256];
  __shared__ int nbs[32];
  int lane = threadIdx.x;
  int b = blockIdx.x / NPTS;
  int n = blockIdx.x % NPTS;
  int C2 = 2 * Cin;
  int base = b * bs;
  int c2sh = __builtin_ctz(C2P);            // C2P is a power of two
  const int* ip = idx + (b * NPTS + n) * KNN;
  nbs[lane] = ip[lane < KNN ? lane : 0];
  __syncthreads();

  for (int i = lane; i < 32 * C2P; i += 32) {
    int mrow = i >> c2sh, c = i & (C2P - 1);
    float v;
    if (c < Cin)     v = ldE(in, base + nbs[mrow] * ns + c * cs, bf) -
                         ldE(in, base + n * ns + c * cs, bf);
    else if (c < C2) v = ldE(in, base + n * ns + (c - Cin) * cs, bf);
    else             v = 0.f;
    feat[mrow][c] = (__bf16)v;
  }

  int col = lane & 15, half = lane >> 4;
  for (int ot = 0; ot < Oc; ot += 16) {
    __syncthreads();
    for (int i = lane; i < 16 * C2P; i += 32) {
      int row = i >> c2sh, k = i & (C2P - 1);
      wtile[row][k] = (__bf16)((k < C2) ? w[(ot + row) * C2 + k] : 0.f);
    }
    __syncthreads();

    int o = ot + col;
    v8f acc0 = {}, acc1 = {};
    for (int kc = 0; kc < C2P; kc += 32) {
      v16bf a0, a1, wf;
#pragma unroll
      for (int j = 0; j < 8; ++j) {
        int kb = kc + ((j >> 2) << 4) + ((j & 3) << 1) + (half << 3);
        a0[2*j]   = feat[col][kb];
        a0[2*j+1] = feat[col][kb+1];
        a1[2*j]   = feat[16 + col][kb];
        a1[2*j+1] = feat[16 + col][kb+1];
        wf[2*j]   = wtile[col][kb];
        wf[2*j+1] = wtile[col][kb+1];
      }
      acc0 = __builtin_amdgcn_wmma_f32_16x16x32_bf16(false, a0, false, wf,
                                                     (short)0, acc0, false, false);
      acc1 = __builtin_amdgcn_wmma_f32_16x16x32_bf16(false, a1, false, wf,
                                                     (short)0, acc1, false, false);
    }
    float sv = sc[o], bv = bi[o];
    float mx = -3.0e38f;
#pragma unroll
    for (int j = 0; j < 8; ++j) {
      float h0 = fmaf(acc0[j], sv, bv); h0 = h0 > 0.f ? h0 : 0.2f * h0;
      float h1 = fmaf(acc1[j], sv, bv); h1 = h1 > 0.f ? h1 : 0.2f * h1;
      mx = fmaxf(mx, fmaxf(h0, h1));
    }
    mx = fmaxf(mx, __shfl_xor(mx, 16));
    if (lane < 16)
      xcat[(b * NPTS + n) * 512 + out_off + o] = (__bf16)mx;
  }
}

// ---------------------------------------------------------------------------
// xcat @ w5^T with fused leaky-relu + max/mean pooling over N.
// Weights staged once per block; activation tiles double-buffered via async
// global->LDS copies overlapping the WMMA loop. (B,N,1024) never exists.
// ---------------------------------------------------------------------------
__global__ __launch_bounds__(32)
void gemm_pool_kernel(const __bf16* __restrict__ xcat, const float* __restrict__ w5,
                      const float* __restrict__ s5, const float* __restrict__ b5,
                      float* __restrict__ pooled) {
  __shared__ __align__(16) __bf16 atile[2][16][512];
  __shared__ __align__(16) __bf16 wtile[16][512];
  int lane = threadIdx.x;
  int b  = blockIdx.x / 64;
  int ot = blockIdx.x % 64;
  int col = lane & 15, half = lane >> 4;
  int o = ot * 16 + col;
  const int NT = NPTS / 16;

  for (int i = lane; i < 16 * 512; i += 32) {
    int row = i >> 9, c = i & 511;
    wtile[row][c] = (__bf16)w5[(ot * 16 + row) * 512 + c];
  }
#if HAVE_ASYNC_LDS
  for (int ch = lane; ch < 16 * 64; ch += 32) {          // tile 0: 64 x 16B / row
    int m = ch >> 6, c0 = (ch & 63) << 3;
    async_copy16(xcat + (size_t)(b * NPTS + m) * 512 + c0, &atile[0][m][c0]);
  }
#endif
  __syncthreads();

  float sv = s5[o], bv = b5[o];
  float runmax = -3.0e38f, runsum = 0.f;

  for (int nt = 0; nt < NT; ++nt) {
    const __bf16 (*at)[512];
#if HAVE_ASYNC_LDS
    async_wait();                   // tile nt resident
    __syncthreads();                // prior readers of buffer (nt+1)&1 done
    if (nt + 1 < NT) {
      int nb = (nt + 1) & 1;
      for (int ch = lane; ch < 16 * 64; ch += 32) {
        int m = ch >> 6, c0 = (ch & 63) << 3;
        async_copy16(xcat + (size_t)(b * NPTS + (nt + 1) * 16 + m) * 512 + c0,
                     &atile[nb][m][c0]);
      }
    }
    at = atile[nt & 1];
#else
    __syncthreads();
    for (int i = lane; i < 16 * 512; i += 32) {
      int m = i >> 9, c = i & 511;
      atile[0][m][c] = xcat[(size_t)(b * NPTS + nt * 16 + m) * 512 + c];
    }
    __syncthreads();
    at = atile[0];
#endif

    v8f acc = {};
    for (int kc = 0; kc < 512; kc += 32) {
      v16bf a, wf;
#pragma unroll
      for (int j = 0; j < 8; ++j) {
        int kb = kc + ((j >> 2) << 4) + ((j & 3) << 1) + (half << 3);
        a[2*j]    = at[col][kb];
        a[2*j+1]  = at[col][kb + 1];
        wf[2*j]   = wtile[col][kb];
        wf[2*j+1] = wtile[col][kb + 1];
      }
      acc = __builtin_amdgcn_wmma_f32_16x16x32_bf16(false, a, false, wf,
                                                    (short)0, acc, false, false);
    }
#pragma unroll
    for (int j = 0; j < 8; ++j) {
      float h = fmaf(acc[j], sv, bv); h = h > 0.f ? h : 0.2f * h;
      runmax = fmaxf(runmax, h);
      runsum += h;
    }
  }
  runmax = fmaxf(runmax, __shfl_xor(runmax, 16));
  runsum += __shfl_xor(runsum, 16);
  if (lane < 16) {
    pooled[b * 2048 + o]        = runmax;
    pooled[b * 2048 + 1024 + o] = runsum * (1.f / NPTS);
  }
}

// ---------------------------------------------------------------------------
// Tiny FC head (B=8, <20 MFLOP): one thread per (b,o).
// ---------------------------------------------------------------------------
__global__ void fc_kernel(const float* __restrict__ in, const float* __restrict__ w,
                          const float* __restrict__ sc, const float* __restrict__ bi,
                          float* __restrict__ out, int IN, int OUT, int do_relu) {
  int i = blockIdx.x * blockDim.x + threadIdx.x;
  if (i >= BATCH * OUT) return;
  int b = i / OUT, o = i % OUT;
  const float* ip = in + b * IN;
  const float* wp = w + o * IN;
  float s = 0.f;
  for (int c = 0; c < IN; ++c) s = fmaf(ip[c], wp[c], s);
  float h = s * (sc ? sc[o] : 1.f) + bi[o];
  out[i] = (do_relu && h < 0.f) ? 0.f : h;
}

extern "C" void kernel_launch(void* const* d_in, const int* in_sizes, int n_in,
                              void* d_out, int out_size, void* d_ws, size_t ws_size,
                              hipStream_t stream) {
  const float* x     = (const float*)d_in[0];
  const float* w1    = (const float*)d_in[1];
  const float* s1    = (const float*)d_in[2];
  const float* b1    = (const float*)d_in[3];
  const float* w2    = (const float*)d_in[4];
  const float* s2    = (const float*)d_in[5];
  const float* b2    = (const float*)d_in[6];
  const float* w3    = (const float*)d_in[7];
  const float* s3    = (const float*)d_in[8];
  const float* b3    = (const float*)d_in[9];
  const float* w4    = (const float*)d_in[10];
  const float* s4    = (const float*)d_in[11];
  const float* b4    = (const float*)d_in[12];
  const float* w5    = (const float*)d_in[13];
  const float* s5    = (const float*)d_in[14];
  const float* b5    = (const float*)d_in[15];
  const float* fc1_w = (const float*)d_in[16];
  const float* s6    = (const float*)d_in[17];
  const float* b6    = (const float*)d_in[18];
  const float* fc2_w = (const float*)d_in[19];
  const float* s7    = (const float*)d_in[20];
  const float* b7    = (const float*)d_in[21];
  const float* fc3_w = (const float*)d_in[22];
  const float* fc3_b = (const float*)d_in[23];

  char* ws = (char*)d_ws;
  size_t off = 0;
  __bf16* xcat  = (__bf16*)(ws + off); off += (size_t)BATCH * NPTS * 512 * 2;  // 16 MB
  float* xx     = (float*)(ws + off);  off += (size_t)BATCH * NPTS * 4;
  int*   idx    = (int*)  (ws + off);  off += (size_t)BATCH * NPTS * KNN * 4;
  float* pooled = (float*)(ws + off);  off += (size_t)BATCH * 2048 * 4;
  float* h1     = (float*)(ws + off);  off += (size_t)BATCH * 512 * 4;
  float* h2     = (float*)(ws + off);  off += (size_t)BATCH * 256 * 4;
  (void)ws_size; (void)in_sizes; (void)n_in; (void)out_size;

  const int nknn = BATCH * (NPTS / 16);
  const int nec  = BATCH * NPTS;
  const int nsq  = (BATCH * NPTS + 255) / 256;
  const int S512 = NPTS * 512;

  // ---- Layer 1: input x (B,3,N) f32: elem(b,n,c) = x[b*3N + c*N + n]
  sqnorm_kernel   <<<nsq, 256, 0, stream>>>(x, 0, 3 * NPTS, 1, NPTS, 3, xx);
  knn_topk_kernel <<<nknn, 32, 0, stream>>>(x, 0, 3 * NPTS, 1, NPTS, 3, 32, xx, idx);
  edge_conv_kernel<<<nec, 32, 0, stream>>>(x, 0, 3 * NPTS, 1, NPTS, 3, 32, 64,
                                           w1, s1, b1, idx, xcat, 0);
  // ---- Layer 2: input = xcat[:, :, 0:64] bf16
  sqnorm_kernel   <<<nsq, 256, 0, stream>>>(xcat, 1, S512, 512, 1, 64, xx);
  knn_topk_kernel <<<nknn, 32, 0, stream>>>(xcat, 1, S512, 512, 1, 64, 64, xx, idx);
  edge_conv_kernel<<<nec, 32, 0, stream>>>(xcat, 1, S512, 512, 1, 64, 128, 64,
                                           w2, s2, b2, idx, xcat, 64);
  // ---- Layer 3: input = xcat[:, :, 64:128]
  sqnorm_kernel   <<<nsq, 256, 0, stream>>>(xcat + 64, 1, S512, 512, 1, 64, xx);
  knn_topk_kernel <<<nknn, 32, 0, stream>>>(xcat + 64, 1, S512, 512, 1, 64, 64, xx, idx);
  edge_conv_kernel<<<nec, 32, 0, stream>>>(xcat + 64, 1, S512, 512, 1, 64, 128, 128,
                                           w3, s3, b3, idx, xcat, 128);
  // ---- Layer 4: input = xcat[:, :, 128:256]
  sqnorm_kernel   <<<nsq, 256, 0, stream>>>(xcat + 128, 1, S512, 512, 1, 128, xx);
  knn_topk_kernel <<<nknn, 32, 0, stream>>>(xcat + 128, 1, S512, 512, 1, 128, 128, xx, idx);
  edge_conv_kernel<<<nec, 32, 0, stream>>>(xcat + 128, 1, S512, 512, 1, 128, 256, 256,
                                           w4, s4, b4, idx, xcat, 256);

  // ---- w5 GEMM + fused max/mean pooling
  gemm_pool_kernel<<<BATCH * 64, 32, 0, stream>>>(xcat, w5, s5, b5, pooled);

  // ---- FC head
  fc_kernel<<<(BATCH * 512 + 255) / 256, 256, 0, stream>>>(pooled, fc1_w, s6, b6, h1, 2048, 512, 1);
  fc_kernel<<<(BATCH * 256 + 255) / 256, 256, 0, stream>>>(h1, fc2_w, s7, b7, h2, 512, 256, 1);
  fc_kernel<<<(BATCH * 40 + 255) / 256, 256, 0, stream>>>(h2, fc3_w, nullptr, fc3_b,
                                                          (float*)d_out, 256, 40, 0);
}
